// TSDCD_66924180407085
// MI455X (gfx1250) — matrix operand order repaired and checked by reading, not compiled, not run on hardware
//
#include <hip/hip_runtime.h>

// MI455X / gfx1250, wave32. Fused masked-window MLP, v_wmma_f32_16x16x32_f16.
// 8 waves/block: 4 h-tiles x 2-way K-split. Two k-chunks per barrier stage so
// every wave does one unconditional WMMA per stage (no exec-mask churn).

typedef __attribute__((ext_vector_type(16))) _Float16 v16h;
typedef __attribute__((ext_vector_type(8)))  _Float16 v8h;
typedef __attribute__((ext_vector_type(8)))  float    v8f;

#define B_        16
#define TAU_      10
#define D_        6
#define N_        512
#define W_        9
#define HID_      64
#define K_        540      // TAU_*D_*W_
#define NST       9        // stages; 18 chunks of 32 (K padded 540 -> 576)
#define TPB_TILES 8        // 16-row tiles per block; 512 tiles per i

__device__ __forceinline__ v16h make_afrag(const _Float16* rowp, int lhi) {
  // A 16x32 fragment, contiguous K: lane holds K = lhi*16 + e (matches B fill).
  v8h lo = *(const v8h*)(rowp + lhi * 16);
  v8h hi = *(const v8h*)(rowp + lhi * 16 + 8);
  v16h a;
#pragma unroll
  for (int e = 0; e < 8; ++e) { a[e] = lo[e]; a[e + 8] = hi[e]; }
  return a;
}

__launch_bounds__(256)
__global__ void tsdcd_fused(const float* __restrict__ x,    // (B,TAU,D,N)
                            const float* __restrict__ mp,   // (TAU,D,D,W)
                            const float* __restrict__ u,    // (N,B,TAU,D,D,W)
                            const float* __restrict__ W0,   // (D,540,64)
                            const float* __restrict__ b0,   // (D,64)
                            const float* __restrict__ W1,   // (D,64,64)
                            const float* __restrict__ b1,   // (D,64)
                            const float* __restrict__ W2,   // (D,64,2)
                            const float* __restrict__ b2,   // (D,2)
                            float* __restrict__ out) {      // (B,D,N,2)
  const int i    = blockIdx.y;     // channel 0..5
  const int tid  = threadIdx.x;    // 256 threads = 8 waves
  const int lane = tid & 31;
  const int wv   = tid >> 5;       // 0..7
  const int g    = wv >> 2;        // K-split group: consumes chunk 2s+g
  const int ht   = wv & 3;         // h-tile: HID cols [ht*16, ht*16+16)
  const int lhi  = lane >> 4;
  const int llo  = lane & 15;
  const int kk   = lane;           // m-tile column owned by this thread

  __shared__ __align__(32) _Float16 At[2][16][64];   // double-buffered 2-chunk stage
  __shared__ __align__(32) _Float16 H0[16][HID_];
  __shared__ __align__(32) _Float16 H1[16][HID_];
  __shared__ float Hred[16][HID_];                   // cross-group C reduction
  __shared__ float expw[K_];                         // exp(-mask_param[t,i,j,w])

  // sigmoid(p + log(u/(1-u))) == u / (u + (1-u)*exp(-p))
  for (int k = tid; k < K_; k += 256) {
    int t = k / 54, rem = k - t * 54, j = rem / 9, w = rem - j * 9;
    expw[k] = __expf(-mp[((t * D_ + i) * D_ + j) * W_ + w]);
  }

  // Register-resident B fragments (this kernel's contiguous-K B 32x16 order:
  // lane holds col n = lane%16, K = (lane/16)*16 + e). Wave owns chunks
  // kc = 2*ci + g; for g=1, ci=8 -> chunk 17 is all-zero padding.
  const int ncol = ht * 16 + llo;
  v16h bf0[NST];
#pragma unroll
  for (int ci = 0; ci < NST; ++ci) {
    const int kc = 2 * ci + g;
    v16h b;
#pragma unroll
    for (int e = 0; e < 16; ++e) {
      int k = kc * 32 + lhi * 16 + e;
      b[e] = (_Float16)((k < K_) ? W0[(i * K_ + k) * HID_ + ncol] : 0.0f);
    }
    bf0[ci] = b;
  }
  v16h bf1[2];
#pragma unroll
  for (int kc = 0; kc < 2; ++kc) {
    v16h b;
#pragma unroll
    for (int e = 0; e < 16; ++e) {
      int h = kc * 32 + lhi * 16 + e;
      b[e] = (_Float16)W1[(i * HID_ + h) * HID_ + ncol];
    }
    bf1[kc] = b;
  }
  const float b0v = b0[i * HID_ + ncol];
  const float b1v = b1[i * HID_ + ncol];
  __syncthreads();

  const int tile0 = blockIdx.x * TPB_TILES;
  for (int tt = 0; tt < TPB_TILES; ++tt) {
    const int rbase = (tile0 + tt) * 16;       // rows r = b*512 + n
    const int bb = rbase >> 9;                 // uniform within tile
    const int n0 = rbase & (N_ - 1);
    // flat 32-bit element offsets -> global_load saddr + voffset (scale_offset)
    const uint32_t ubase = (uint32_t)((n0 * B_ + bb) * 3240 + i * 54);
    const uint32_t xbase = (uint32_t)(bb * (TAU_ * D_ * N_));

    v8f c0 = {};
#pragma unroll
    for (int s = 0; s < NST; ++s) {
      // build chunks 2s (cols 0..31) and 2s+1 (cols 32..63) of this stage
#pragma unroll
      for (int half = 0; half < 2; ++half) {
        const int k = (2 * s + half) * 32 + kk;
#pragma unroll
        for (int q = 0; q < 2; ++q) {
          const int row = wv + q * 8;          // this thread's 2 rows
          float mval = 0.0f;
          if (k < K_) {                        // folds to true except stage 8
            int t = k / 54, rem = k - t * 54;
            int j = rem / 9, w = rem - j * 9;
            float uu  = u[ubase + (uint32_t)row * 51840u
                          + (uint32_t)(t * 324 + rem)];
            float den = fmaf(1.0f - uu, expw[k], uu);
            float msk = uu * __builtin_amdgcn_rcpf(den);
            int xi  = n0 + row + w - 4;
            int xic = min(max(xi, 0), N_ - 1); // branchless bounds
            float xv = x[xbase + (uint32_t)((t * D_ + j) * N_ + xic)];
            mval = (xi == xic) ? xv * msk : 0.0f;
          }
          At[s & 1][row][half * 32 + kk] = (_Float16)mval;
        }
      }
      __syncthreads();
      // every wave: one unconditional WMMA on its chunk 2s+g
      v16h a = make_afrag(&At[s & 1][llo][g * 32], lhi);
      c0 = __builtin_amdgcn_wmma_f32_16x16x32_f16(
              false, a, false, bf0[s], (short)0, c0, false, false);
    }

    // combine the two K-split partials, then bias+relu -> H0 (f16)
    if (g == 1) {
#pragma unroll
      for (int v = 0; v < 8; ++v) Hred[v + 8 * lhi][ncol] = c0[v];
    }
    __syncthreads();
    if (g == 0) {
#pragma unroll
      for (int v = 0; v < 8; ++v) {
        float hv = c0[v] + Hred[v + 8 * lhi][ncol] + b0v;
        H0[v + 8 * lhi][ncol] = (_Float16)fmaxf(hv, 0.0f);
      }
    }
    __syncthreads();

    // layer 1: 16x64 @ 64x64 on group 0 waves
    if (g == 0) {
      v8f c1 = {};
#pragma unroll
      for (int kc = 0; kc < 2; ++kc) {
        v16h a = make_afrag(&H0[llo][kc * 32], lhi);
        c1 = __builtin_amdgcn_wmma_f32_16x16x32_f16(
                false, a, false, bf1[kc], (short)0, c1, false, false);
      }
#pragma unroll
      for (int v = 0; v < 8; ++v)
        H1[v + 8 * lhi][ncol] = (_Float16)fmaxf(c1[v] + b1v, 0.0f);
    }
    __syncthreads();

    // layer 2: 64 -> 2, scalar dots on wave 0
    if (tid < 32) {
      int row = tid >> 1, o = tid & 1;
      float acc = b2[i * 2 + o];
#pragma unroll
      for (int gg = 0; gg < HID_; ++gg)
        acc += (float)H1[row][gg] * W2[(i * HID_ + gg) * 2 + o];
      int n = n0 + row;
      out[((bb * D_ + i) * N_ + n) * 2 + o] = acc;
    }
    __syncthreads();
  }
}

extern "C" void kernel_launch(void* const* d_in, const int* in_sizes, int n_in,
                              void* d_out, int out_size, void* d_ws, size_t ws_size,
                              hipStream_t stream) {
  (void)in_sizes; (void)n_in; (void)out_size; (void)d_ws; (void)ws_size;
  const float* x  = (const float*)d_in[0];
  const float* mp = (const float*)d_in[1];
  const float* u  = (const float*)d_in[2];
  const float* W0 = (const float*)d_in[3];
  const float* b0 = (const float*)d_in[4];
  const float* W1 = (const float*)d_in[5];
  const float* b1 = (const float*)d_in[6];
  const float* W2 = (const float*)d_in[7];
  const float* b2 = (const float*)d_in[8];
  float* out = (float*)d_out;

  dim3 grid((B_ * N_ / 16) / TPB_TILES, D_);  // (64, 6)
  tsdcd_fused<<<grid, 256, 0, stream>>>(x, mp, u, W0, b0, W1, b1, W2, b2, out);
}
